// SpecTNT_CAMHSA_68058051772575
// MI455X (gfx1250) — matrix operand order repaired and checked by reading, not compiled, not run on hardware
//
#include <hip/hip_runtime.h>

// ---------------------------------------------------------------------------
// SpecTNT block for MI455X (gfx1250): bf16 WMMA (v_wmma_f32_16x16x32_bf16),
// wave32 shuffles for softmax/LN, LDS-staged tiles with b128-wide traffic.
// GEMM uses double-buffered LDS so global_load_b128 for tile k+1 issues
// before the WMMA on tile k (latency hidden under the matrix op).
// ---------------------------------------------------------------------------

typedef __bf16 bf16;
typedef __attribute__((ext_vector_type(16))) __bf16 v16bf;
typedef __attribute__((ext_vector_type(8)))  __bf16 v8bf;
typedef __attribute__((ext_vector_type(8)))  float  v8f;

#define NSEQ   2048        // B*T
#define FTOK   93
#define FP     96          // padded F
#define DD     128
#define MS     (NSEQ * FTOK)   // 190464 spectral rows (mult of 64)
#define POSC   539

union AccF  { v8f v; float e[8]; };
union Frag16 { v16bf v; v8bf h[2]; bf16 e[16]; };
union Pack8 { v8bf v; bf16 e[8]; };
union Pack4 { unsigned long long u; bf16 e[4]; };

__device__ __forceinline__ v8f vzero8() {
  v8f z = {0.f, 0.f, 0.f, 0.f, 0.f, 0.f, 0.f, 0.f};
  return z;
}

__device__ __forceinline__ v8bf zero8bf() {
  Pack8 p;
#pragma unroll
  for (int i = 0; i < 8; ++i) p.e[i] = (bf16)0.f;
  return p.v;
}

__device__ __forceinline__ v8bf pack8(const float4& a, const float4& b) {
  Pack8 p;
  p.e[0] = (bf16)a.x; p.e[1] = (bf16)a.y; p.e[2] = (bf16)a.z; p.e[3] = (bf16)a.w;
  p.e[4] = (bf16)b.x; p.e[5] = (bf16)b.y; p.e[6] = (bf16)b.z; p.e[7] = (bf16)b.w;
  return p.v;
}

__device__ __forceinline__ void pack4_store(bf16* dst, const float4& a) {
  Pack4 p;
  p.e[0] = (bf16)a.x; p.e[1] = (bf16)a.y; p.e[2] = (bf16)a.z; p.e[3] = (bf16)a.w;
  *(unsigned long long*)dst = p.u;   // 8B-aligned ds_store_b64
}

__device__ __forceinline__ v8f wmma_bf16(v16bf a, v16bf b, v8f c) {
  return __builtin_amdgcn_wmma_f32_16x16x32_bf16(
      false, a, false, b, (short)0, c, false, false);
}

// A fragment: 16x32 bf16 tile, row stride ld (mult of 8). Two b128 loads.
__device__ __forceinline__ v16bf frag_a(const bf16* p, int ld) {
  int lane = threadIdx.x & 31;
  const bf16* r = p + (lane & 15) * ld + ((lane >> 4) << 3);
  Frag16 f;
  f.h[0] = *(const v8bf*)(r);
  f.h[1] = *(const v8bf*)(r + 16);
  return f.v;
}

// B fragment from N-major buffer Bt[n][k], row stride ld (mult of 8).
__device__ __forceinline__ v16bf frag_bt(const bf16* p, int ld) {
  int lane = threadIdx.x & 31;
  const bf16* r = p + (lane & 15) * ld + ((lane >> 4) << 3);
  Frag16 f;
  f.h[0] = *(const v8bf*)(r);
  f.h[1] = *(const v8bf*)(r + 16);
  return f.v;
}

// ---------------------------------------------------------------------------
// GEMM: C[M,N] = A[M,K] @ W[K,N] + bias (+ residual).
// Requirements (all call sites satisfy): M mult of 64, N/K mult of 32.
// 8 waves, block tile 64x32, one 16x16 WMMA tile per wave, double-buffered
// LDS: one barrier per k-step, next-tile global loads overlap the WMMA.
// ---------------------------------------------------------------------------
template <bool RES>
__global__ __launch_bounds__(256) void gemm_bias_kernel(
    const float* __restrict__ A, const float* __restrict__ W,
    const float* __restrict__ bias, const float* __restrict__ residual,
    float* __restrict__ C, int M, int N, int K) {
  __shared__ bf16 As[2][64 * 32];    // [m][k]
  __shared__ bf16 WsT[2][32 * 32];   // [n][k] transposed
  const int m0 = blockIdx.x * 64;
  const int n0 = blockIdx.y * 32;
  const int tid = threadIdx.x;
  const int wave = tid >> 5;
  const int wm = (wave >> 1) * 16;
  const int wn = (wave & 1) * 16;
  AccF acc; acc.v = vzero8();

  const int ar = tid >> 2;            // 0..63
  const int ac = (tid & 3) * 8;       // 0,8,16,24
  const int wr = tid >> 3;            // 0..31
  const int wc = (tid & 7) * 4;       // 0..28
  const float* Arow = A + (size_t)(m0 + ar) * K + ac;
  const float* Wrow = W + (size_t)wr * N + n0 + wc;

  // stage tile 0
  float4 a0 = *(const float4*)(Arow);
  float4 a1 = *(const float4*)(Arow + 4);
  float4 w  = *(const float4*)(Wrow);
  *(v8bf*)&As[0][ar * 32 + ac] = pack8(a0, a1);
  WsT[0][(wc + 0) * 32 + wr] = (bf16)w.x;
  WsT[0][(wc + 1) * 32 + wr] = (bf16)w.y;
  WsT[0][(wc + 2) * 32 + wr] = (bf16)w.z;
  WsT[0][(wc + 3) * 32 + wr] = (bf16)w.w;
  __syncthreads();

  int cur = 0;
  for (int kb = 0; kb < K; kb += 32) {
    const bool hasNext = (kb + 32 < K);
    if (hasNext) {   // issue next-tile loads before consuming current tile
      a0 = *(const float4*)(Arow + kb + 32);
      a1 = *(const float4*)(Arow + kb + 36);
      w  = *(const float4*)(Wrow + (size_t)(kb + 32) * N);
    }
    v16bf a = frag_a(&As[cur][wm * 32], 32);
    v16bf b = frag_bt(&WsT[cur][wn * 32], 32);
    acc.v = wmma_bf16(a, b, acc.v);
    if (hasNext) {
      int nxt = cur ^ 1;
      *(v8bf*)&As[nxt][ar * 32 + ac] = pack8(a0, a1);
      WsT[nxt][(wc + 0) * 32 + wr] = (bf16)w.x;
      WsT[nxt][(wc + 1) * 32 + wr] = (bf16)w.y;
      WsT[nxt][(wc + 2) * 32 + wr] = (bf16)w.z;
      WsT[nxt][(wc + 3) * 32 + wr] = (bf16)w.w;
      __syncthreads();
      cur = nxt;
    }
  }

  const int lane = tid & 31;
  const int n = lane & 15, mb = (lane >> 4) << 3;
#pragma unroll
  for (int i = 0; i < 8; ++i) {
    int gm = m0 + wm + mb + i;
    int gn = n0 + wn + n;
    float v = acc.e[i] + bias[gn];
    if (RES) v += residual[(size_t)gm * N + gn];
    C[(size_t)gm * N + gn] = v;
  }
}

// ---------------------------------------------------------------------------
// Row LayerNorm, d=128. One wave per row; b128 row I/O.
// ---------------------------------------------------------------------------
__global__ __launch_bounds__(256) void layernorm128_kernel(
    const float* __restrict__ X, const float* __restrict__ gamma,
    const float* __restrict__ beta, float* __restrict__ Y, int M) {
  int row = blockIdx.x * 8 + (threadIdx.x >> 5);
  if (row >= M) return;
  int lane = threadIdx.x & 31;
  const float4 xv = *(const float4*)(X + (size_t)row * 128 + lane * 4);
  float v[4] = {xv.x, xv.y, xv.z, xv.w};
  float s = v[0] + v[1] + v[2] + v[3];
  for (int o = 16; o; o >>= 1) s += __shfl_xor(s, o, 32);
  float mean = s * (1.f / 128.f);
  float q = 0.f;
#pragma unroll
  for (int i = 0; i < 4; ++i) { float d = v[i] - mean; q += d * d; }
  for (int o = 16; o; o >>= 1) q += __shfl_xor(q, o, 32);
  float rstd = rsqrtf(q * (1.f / 128.f) + 1e-7f);
  const float4 g4 = *(const float4*)(gamma + lane * 4);
  const float4 b4 = *(const float4*)(beta + lane * 4);
  float4 o4;
  o4.x = g4.x * (v[0] - mean) * rstd + b4.x;
  o4.y = g4.y * (v[1] - mean) * rstd + b4.y;
  o4.z = g4.z * (v[2] - mean) * rstd + b4.z;
  o4.w = g4.w * (v[3] - mean) * rstd + b4.w;
  *(float4*)(Y + (size_t)row * 128 + lane * 4) = o4;
}

// ---------------------------------------------------------------------------
// Spectral attention: grid (2048 seq, 4 heads), dh=32, F padded 93->96.
// Mask is evaluated branchlessly (pure function of indices -> v_cndmask).
// ---------------------------------------------------------------------------
__global__ __launch_bounds__(256) void spec_attn_kernel(
    const float* __restrict__ Q, const float* __restrict__ K,
    const float* __restrict__ V, float* __restrict__ O) {
  extern __shared__ char smem[];
  bf16*  Qs  = (bf16*)smem;                // 96 x 32 [f][d]
  bf16*  Ks  = Qs + FP * 32;               // 96 x 32 [f][d]  (B^T for scores)
  bf16*  VsT = Ks + FP * 32;               // 32 x 96 [d][f]  (B^T for P@V)
  float* S   = (float*)(VsT + 32 * FP);    // 96 x 96 f32
  bf16*  P   = (bf16*)(S + FP * FP);       // 96 x 96 bf16

  const int s = blockIdx.x, h = blockIdx.y;
  const int tid = threadIdx.x, wave = tid >> 5;
  const size_t base = (size_t)s * FTOK * DD + h * 32;

#pragma unroll
  for (int it = 0; it < 3; ++it) {
    int lin = it * 256 + tid;           // 0..767 float4 groups
    int r = lin >> 3;                   // 0..95
    int c4 = (lin & 7) * 4;             // 0..28
    float4 q4 = make_float4(0.f, 0.f, 0.f, 0.f), k4 = q4, v4 = q4;
    if (r < FTOK) {
      size_t g = base + (size_t)r * DD + c4;
      q4 = *(const float4*)(Q + g);
      k4 = *(const float4*)(K + g);
      v4 = *(const float4*)(V + g);
    }
    pack4_store(&Qs[r * 32 + c4], q4);
    pack4_store(&Ks[r * 32 + c4], k4);
    VsT[(c4 + 0) * FP + r] = (bf16)v4.x;
    VsT[(c4 + 1) * FP + r] = (bf16)v4.y;
    VsT[(c4 + 2) * FP + r] = (bf16)v4.z;
    VsT[(c4 + 3) * FP + r] = (bf16)v4.w;
  }
  __syncthreads();

  // scores: 6x6 tiles, contraction dh=32 (1 WMMA each) + mask + scale
  for (int t = wave; t < 36; t += 8) {
    int ti = t / 6, tj = t % 6;
    v16bf a = frag_a(&Qs[ti * 16 * 32], 32);
    v16bf b = frag_bt(&Ks[tj * 16 * 32], 32);
    AccF acc; acc.v = wmma_bf16(a, b, vzero8());
    int lane = tid & 31, n = lane & 15, mb = (lane >> 4) << 3;
    int cm = tj * 16 + n;
    // column-only predicates (loop-invariant over i)
    int c_in   = (cm < FTOK);
    int c_nz   = (cm != 0);
    int c_mel  = (cm <= 80);
#pragma unroll
    for (int i = 0; i < 8; ++i) {
      int rn = ti * 16 + mb + i;
      // branchless spectral mask: row0 sees all; rows>=1 block-diagonal,
      // never column 0; padded rows/cols masked.
      int ok = ((rn == 0) | (c_nz & ((rn <= 80) == c_mel)))
             & c_in & (rn < FTOK);
      S[rn * FP + cm] = ok ? acc.e[i] * 0.1767766953f : -1e12f;
    }
  }
  __syncthreads();

  // softmax per row; probs -> bf16
  for (int r = wave; r < FP; r += 8) {
    int lane = tid & 31;
    if (r < FTOK) {
      float mx = -1e30f;
#pragma unroll
      for (int i = 0; i < 3; ++i) mx = fmaxf(mx, S[r * FP + lane + 32 * i]);
      for (int o = 16; o; o >>= 1) mx = fmaxf(mx, __shfl_xor(mx, o, 32));
      float e[3], sum = 0.f;
#pragma unroll
      for (int i = 0; i < 3; ++i) {
        e[i] = __expf(S[r * FP + lane + 32 * i] - mx);
        sum += e[i];
      }
      for (int o = 16; o; o >>= 1) sum += __shfl_xor(sum, o, 32);
      float inv = 1.f / sum;
#pragma unroll
      for (int i = 0; i < 3; ++i) P[r * FP + lane + 32 * i] = (bf16)(e[i] * inv);
    } else {
#pragma unroll
      for (int i = 0; i < 3; ++i) P[r * FP + lane + 32 * i] = (bf16)0.f;
    }
  }
  __syncthreads();

  // O = P(96x96) @ V(96x32): 6 M-tiles x 2 N-tiles, K=96 (3 steps)
  for (int t = wave; t < 12; t += 8) {
    int ti = t >> 1, tj = t & 1;
    AccF acc; acc.v = vzero8();
#pragma unroll
    for (int ks = 0; ks < 3; ++ks) {
      v16bf a = frag_a(&P[ti * 16 * FP + ks * 32], FP);
      v16bf b = frag_bt(&VsT[(tj * 16) * FP + ks * 32], FP);
      acc.v = wmma_bf16(a, b, acc.v);
    }
    int lane = tid & 31, n = lane & 15, mb = (lane >> 4) << 3;
#pragma unroll
    for (int i = 0; i < 8; ++i) {
      int f = ti * 16 + mb + i;
      if (f < FTOK) O[base + (size_t)f * DD + tj * 16 + n] = acc.e[i];
    }
  }
}

// ---------------------------------------------------------------------------
// Fused FFN + residual + LayerNorm: Y = LN(relu(X@W1+b1)@W2 + b2 + X).
// 16-row blocks; hidden (16x512) kept in LDS bf16; weights staged transposed.
// ---------------------------------------------------------------------------
__global__ __launch_bounds__(256) void ffn_kernel(
    const float* __restrict__ X, const float* __restrict__ W1,
    const float* __restrict__ b1, const float* __restrict__ W2,
    const float* __restrict__ b2, const float* __restrict__ gamma,
    const float* __restrict__ beta, float* __restrict__ Y, int M) {
  extern __shared__ char smem[];
  float* Xf  = (float*)smem;               // 16x128 f32
  bf16*  Xb  = (bf16*)(Xf + 16 * 128);     // 16x128 bf16
  bf16*  H   = Xb + 16 * 128;              // 16x512 bf16 [m][k]
  bf16*  WtT = H + 16 * 512;               // [n][k] slab, up to 512x32
  float* Of  = (float*)(WtT + 512 * 32);   // 16x128 f32

  const int m0 = blockIdx.x * 16;
  const int tid = threadIdx.x, wave = tid >> 5;

#pragma unroll
  for (int it = 0; it < 2; ++it) {
    int lin = it * 256 + tid;            // 0..511 float4 groups
    int r = lin >> 5, c4 = (lin & 31) * 4;
    float4 xv = *(const float4*)(X + (size_t)(m0 + r) * 128 + c4);
    *(float4*)&Xf[r * 128 + c4] = xv;
    pack4_store(&Xb[r * 128 + c4], xv);
  }
  __syncthreads();

  // H = relu(X @ W1 + b1): 32 N-tiles (4/wave), K=128 in 4 steps
  AccF hacc[4];
#pragma unroll
  for (int j = 0; j < 4; ++j) hacc[j].v = vzero8();
  for (int ks = 0; ks < 4; ++ks) {
#pragma unroll
    for (int it = 0; it < 16; ++it) {
      int lin = it * 256 + tid;          // 0..4095 float4 groups
      int r = lin >> 7;                  // k row 0..31
      int c4 = (lin & 127) * 4;          // n 0..508
      float4 w = *(const float4*)(W1 + (size_t)(ks * 32 + r) * 512 + c4);
      WtT[(c4 + 0) * 32 + r] = (bf16)w.x;
      WtT[(c4 + 1) * 32 + r] = (bf16)w.y;
      WtT[(c4 + 2) * 32 + r] = (bf16)w.z;
      WtT[(c4 + 3) * 32 + r] = (bf16)w.w;
    }
    __syncthreads();
    v16bf a = frag_a(&Xb[ks * 32], 128);
#pragma unroll
    for (int j = 0; j < 4; ++j) {
      int nt = wave * 4 + j;
      v16bf b = frag_bt(&WtT[(nt * 16) * 32], 32);
      hacc[j].v = wmma_bf16(a, b, hacc[j].v);
    }
    __syncthreads();
  }
  {
    int lane = tid & 31, n = lane & 15, mb = (lane >> 4) << 3;
#pragma unroll
    for (int j = 0; j < 4; ++j) {
      int nt = wave * 4 + j;
#pragma unroll
      for (int i = 0; i < 8; ++i) {
        float v = hacc[j].e[i] + b1[nt * 16 + n];
        H[(mb + i) * 512 + nt * 16 + n] = (bf16)fmaxf(v, 0.f);
      }
    }
  }
  __syncthreads();

  // O = H @ W2 + b2 + X: 8 N-tiles (1/wave), K=512 in 16 steps
  AccF oacc; oacc.v = vzero8();
  for (int ks = 0; ks < 16; ++ks) {
#pragma unroll
    for (int it = 0; it < 4; ++it) {
      int lin = it * 256 + tid;          // 0..1023 float4 groups
      int r = lin >> 5;                  // k row 0..31
      int c4 = (lin & 31) * 4;           // n 0..124
      float4 w = *(const float4*)(W2 + (size_t)(ks * 32 + r) * 128 + c4);
      WtT[(c4 + 0) * 32 + r] = (bf16)w.x;
      WtT[(c4 + 1) * 32 + r] = (bf16)w.y;
      WtT[(c4 + 2) * 32 + r] = (bf16)w.z;
      WtT[(c4 + 3) * 32 + r] = (bf16)w.w;
    }
    __syncthreads();
    v16bf a = frag_a(&H[ks * 32], 512);
    v16bf b = frag_bt(&WtT[(wave * 16) * 32], 32);
    oacc.v = wmma_bf16(a, b, oacc.v);
    __syncthreads();
  }
  {
    int lane = tid & 31, n = lane & 15, mb = (lane >> 4) << 3;
#pragma unroll
    for (int i = 0; i < 8; ++i) {
      int c = wave * 16 + n;
      Of[(mb + i) * 128 + c] = oacc.e[i] + b2[c] + Xf[(mb + i) * 128 + c];
    }
  }
  __syncthreads();

  // LayerNorm rows (2 rows / wave), b128 LDS reads
  for (int r = wave; r < 16; r += 8) {
    int lane = tid & 31;
    const float4 xv = *(const float4*)&Of[r * 128 + lane * 4];
    float v[4] = {xv.x, xv.y, xv.z, xv.w};
    float s = v[0] + v[1] + v[2] + v[3];
    for (int o = 16; o; o >>= 1) s += __shfl_xor(s, o, 32);
    float mean = s * (1.f / 128.f);
    float q = 0.f;
#pragma unroll
    for (int i = 0; i < 4; ++i) { float d = v[i] - mean; q += d * d; }
    for (int o = 16; o; o >>= 1) q += __shfl_xor(q, o, 32);
    float rstd = rsqrtf(q * (1.f / 128.f) + 1e-7f);
    const float4 g4 = *(const float4*)(gamma + lane * 4);
    const float4 b4 = *(const float4*)(beta + lane * 4);
    float4 o4;
    o4.x = g4.x * (v[0] - mean) * rstd + b4.x;
    o4.y = g4.y * (v[1] - mean) * rstd + b4.y;
    o4.z = g4.z * (v[2] - mean) * rstd + b4.z;
    o4.w = g4.w * (v[3] - mean) * rstd + b4.w;
    *(float4*)(Y + (size_t)(m0 + r) * 128 + lane * 4) = o4;
  }
}

// ---------------------------------------------------------------------------
// Temporal attention with relative position bias. grid (32 row-tiles, 8
// heads, 4 batch). dh=16 zero-padded to K=32 for WMMA.
// ---------------------------------------------------------------------------
__global__ __launch_bounds__(256) void temp_attn_kernel(
    const float* __restrict__ Q, const float* __restrict__ K,
    const float* __restrict__ V, const float* __restrict__ pos_k,
    float* __restrict__ O) {
  extern __shared__ char smem[];
  bf16*  Qs  = (bf16*)smem;                 // 16 x 32 [n][d] (d padded)
  bf16*  Ks  = Qs + 16 * 32;                // 512 x 32 [m][d] (B^T for scores)
  bf16*  VsT = Ks + 512 * 32;               // 16 x 512 [d][m] (B^T for P@V)
  float* S   = (float*)(VsT + 16 * 512);    // 16 x 512 f32
  bf16*  P   = (bf16*)(S + 16 * 512);       // 16 x 512 bf16

  const int n0 = blockIdx.x * 16, h = blockIdx.y, b = blockIdx.z;
  const int tid = threadIdx.x, wave = tid >> 5;
  const size_t rowbase = (size_t)b * 512;

#pragma unroll
  for (int it = 0; it < 8; ++it) {
    int lin = it * 256 + tid;               // 0..2047 float4 groups
    int r = lin >> 2;                       // 0..511
    int c4 = (lin & 3) * 4;                 // 0,4,8,12
    size_t g = (rowbase + r) * 128 + h * 16 + c4;
    float4 k4 = *(const float4*)(K + g);
    float4 v4 = *(const float4*)(V + g);
    pack4_store(&Ks[r * 32 + c4], k4);
    VsT[(c4 + 0) * 512 + r] = (bf16)v4.x;
    VsT[(c4 + 1) * 512 + r] = (bf16)v4.y;
    VsT[(c4 + 2) * 512 + r] = (bf16)v4.z;
    VsT[(c4 + 3) * 512 + r] = (bf16)v4.w;
  }
  for (int r = tid; r < 512; r += 256) {    // zero-pad K cols 16..31
    *(v8bf*)&Ks[r * 32 + 16] = zero8bf();
    *(v8bf*)&Ks[r * 32 + 24] = zero8bf();
  }
  if (tid < 64) {
    int r = tid >> 2, c4 = (tid & 3) * 4;
    float4 q4 = *(const float4*)(Q + (rowbase + n0 + r) * 128 + h * 16 + c4);
    pack4_store(&Qs[r * 32 + c4], q4);
  } else if (tid < 96) {
    int rr = (tid - 64) >> 1, half = (tid - 64) & 1;
    *(v8bf*)&Qs[rr * 32 + 16 + half * 8] = zero8bf();
  }
  __syncthreads();

  // raw scores: 32 column tiles, 1 WMMA each
  v16bf aq = frag_a(Qs, 32);
  for (int t = wave; t < 32; t += 8) {
    v16bf bk = frag_bt(&Ks[t * 16 * 32], 32);
    AccF acc; acc.v = wmma_bf16(aq, bk, vzero8());
    int lane = tid & 31, n = lane & 15, mb = (lane >> 4) << 3;
#pragma unroll
    for (int i = 0; i < 8; ++i) S[(mb + i) * 512 + t * 16 + n] = acc.e[i];
  }
  __syncthreads();

  // relative-position bias + 1/sqrt(dh) scale; one q-row per thread
  {
    int r = tid >> 4;                       // 0..15
    int mlane = tid & 15;
    Frag16 qrow;
    qrow.h[0] = *(const v8bf*)&Qs[r * 32];
    qrow.h[1] = *(const v8bf*)&Qs[r * 32 + 8];
    float qv[16];
#pragma unroll
    for (int c = 0; c < 16; ++c) qv[c] = (float)qrow.e[c];
    int n = n0 + r;
    for (int j = 0; j < 32; ++j) {
      int m = mlane + 16 * j;
      int rel = m - n;
      rel = rel < -POSC ? -POSC : (rel > POSC ? POSC : rel);
      const float* pe = pos_k + (size_t)(rel + POSC) * 16;
      float4 p0 = *(const float4*)(pe);
      float4 p1 = *(const float4*)(pe + 4);
      float4 p2 = *(const float4*)(pe + 8);
      float4 p3 = *(const float4*)(pe + 12);
      float d = qv[0] * p0.x + qv[1] * p0.y + qv[2] * p0.z + qv[3] * p0.w
              + qv[4] * p1.x + qv[5] * p1.y + qv[6] * p1.z + qv[7] * p1.w
              + qv[8] * p2.x + qv[9] * p2.y + qv[10] * p2.z + qv[11] * p2.w
              + qv[12] * p3.x + qv[13] * p3.y + qv[14] * p3.z + qv[15] * p3.w;
      S[r * 512 + m] = (S[r * 512 + m] + d) * 0.25f;
    }
  }
  __syncthreads();

  // softmax over 512 (2 rows / wave)
  for (int r = wave; r < 16; r += 8) {
    int lane = tid & 31;
    float mx = -1e30f;
#pragma unroll
    for (int i = 0; i < 16; ++i) mx = fmaxf(mx, S[r * 512 + lane + 32 * i]);
    for (int o = 16; o; o >>= 1) mx = fmaxf(mx, __shfl_xor(mx, o, 32));
    float e[16], sum = 0.f;
#pragma unroll
    for (int i = 0; i < 16; ++i) {
      e[i] = __expf(S[r * 512 + lane + 32 * i] - mx);
      sum += e[i];
    }
    for (int o = 16; o; o >>= 1) sum += __shfl_xor(sum, o, 32);
    float inv = 1.f / sum;
#pragma unroll
    for (int i = 0; i < 16; ++i) P[r * 512 + lane + 32 * i] = (bf16)(e[i] * inv);
  }
  __syncthreads();

  // O = P(16x512) @ V(512x16): 16 k-steps split over 8 waves, LDS reduce
  AccF oacc; oacc.v = vzero8();
  for (int ks = wave; ks < 16; ks += 8) {
    v16bf a = frag_a(&P[ks * 32], 512);
    v16bf bv = frag_bt(&VsT[ks * 32], 512);
    oacc.v = wmma_bf16(a, bv, oacc.v);
  }
  float* R = S;  // reuse score region as 8 x (16x16) partials
  {
    int lane = tid & 31, n = lane & 15, mb = (lane >> 4) << 3;
#pragma unroll
    for (int i = 0; i < 8; ++i) R[wave * 256 + (mb + i) * 16 + n] = oacc.e[i];
  }
  __syncthreads();
  {
    int r = tid >> 4, c = tid & 15;
    float v = 0.f;
#pragma unroll
    for (int w = 0; w < 8; ++w) v += R[w * 256 + r * 16 + c];
    O[(rowbase + n0 + r) * 128 + h * 16 + c] = v;
  }
}

// ---------------------------------------------------------------------------
// Small helpers: gather fct rows; final compose (float4 wide).
// ---------------------------------------------------------------------------
__global__ __launch_bounds__(256) void gather_fct_kernel(
    const float* __restrict__ X2, float* __restrict__ fct) {
  int i = blockIdx.x * 256 + threadIdx.x;       // float4 index
  if (i < NSEQ * 32) {
    int bt = i >> 5, d4 = (i & 31) * 4;
    *(float4*)(fct + (size_t)bt * DD + d4) =
        *(const float4*)(X2 + (size_t)bt * FTOK * DD + d4);
  }
}

__global__ __launch_bounds__(256) void compose_kernel(
    const float* __restrict__ Xs, const float* __restrict__ Fct,
    float* __restrict__ out) {
  size_t i4 = (size_t)blockIdx.x * 256 + threadIdx.x;  // float4 index
  if (i4 >= (size_t)NSEQ * FTOK * 32) return;
  int d = (int)(i4 & 31) * 4;
  size_t rf = i4 >> 5;
  int f = (int)(rf % FTOK);
  size_t bt = rf / FTOK;
  float4 v = (f == 0) ? *(const float4*)(Fct + bt * DD + d)
                      : *(const float4*)(Xs + i4 * 4);
  *(float4*)(out + i4 * 4) = v;
}

// ---------------------------------------------------------------------------
extern "C" void kernel_launch(void* const* d_in, const int* in_sizes, int n_in,
                              void* d_out, int out_size, void* d_ws, size_t ws_size,
                              hipStream_t stream) {
  (void)in_sizes; (void)n_in; (void)out_size; (void)ws_size;
  const float* x = (const float*)d_in[0];
  const float *Wq_f = (const float*)d_in[1],  *bq_f = (const float*)d_in[2];
  const float *Wk_f = (const float*)d_in[3],  *bk_f = (const float*)d_in[4];
  const float *Wv_f = (const float*)d_in[5],  *bv_f = (const float*)d_in[6];
  const float *Wh_f = (const float*)d_in[7],  *bh_f = (const float*)d_in[8];
  const float *ga_f = (const float*)d_in[9],  *be_f = (const float*)d_in[10];
  const float *W1_f = (const float*)d_in[11], *b1_f = (const float*)d_in[12];
  const float *W2_f = (const float*)d_in[13], *b2_f = (const float*)d_in[14];
  const float *gf_f = (const float*)d_in[15], *bf_f = (const float*)d_in[16];
  const float *Wq_t = (const float*)d_in[17], *bq_t = (const float*)d_in[18];
  const float *Wk_t = (const float*)d_in[19], *bk_t = (const float*)d_in[20];
  const float *Wv_t = (const float*)d_in[21], *bv_t = (const float*)d_in[22];
  const float *Wh_t = (const float*)d_in[23], *bh_t = (const float*)d_in[24];
  const float *ga_t = (const float*)d_in[25], *be_t = (const float*)d_in[26];
  const float *pos_k = (const float*)d_in[27];
  const float *W1_t = (const float*)d_in[28], *b1_t = (const float*)d_in[29];
  const float *W2_t = (const float*)d_in[30], *b2_t = (const float*)d_in[31];
  const float *gf_t = (const float*)d_in[32], *bf_t = (const float*)d_in[33];

  float* ws = (float*)d_ws;
  const size_t SZ = (size_t)MS * DD;
  float* Qb = ws;                    // spectral Q (reused: pre-LN)
  float* Kb = ws + SZ;               // spectral K (reused: post-LN X1)
  float* Vb = ws + 2 * SZ;           // spectral V (reused: X2 = spectral out)
  float* Ob = (float*)d_out;         // spectral attn out (scratch until compose)
  float* sm = ws + 3 * SZ;
  const size_t SM = (size_t)NSEQ * DD;
  float* fct  = sm;
  float* tQ   = sm + SM;
  float* tK   = sm + 2 * SM;
  float* tV   = sm + 3 * SM;
  float* tO   = sm + 4 * SM;
  float* tPre = sm + 5 * SM;
  float* tX1  = sm + 6 * SM;
  float* tX2  = sm + 7 * SM;

  const dim3 blk(256);
  const dim3 gS(MS / 64, DD / 32);
  const dim3 gT(NSEQ / 64, DD / 32);
  const size_t lds_spec = (size_t)3 * FP * 32 * 2 + (size_t)FP * FP * 4 + (size_t)FP * FP * 2;
  const size_t lds_ffn  = 16 * 128 * 4 + 16 * 128 * 2 + 16 * 512 * 2 + 512 * 32 * 2 + 16 * 128 * 4;
  const size_t lds_temp = 16 * 32 * 2 + 512 * 32 * 2 + 16 * 512 * 2 + 16 * 512 * 4 + 16 * 512 * 2;

  // ---- spectral transformer ----
  gemm_bias_kernel<false><<<gS, blk, 0, stream>>>(x, Wq_f, bq_f, nullptr, Qb, MS, DD, DD);
  gemm_bias_kernel<false><<<gS, blk, 0, stream>>>(x, Wk_f, bk_f, nullptr, Kb, MS, DD, DD);
  gemm_bias_kernel<false><<<gS, blk, 0, stream>>>(x, Wv_f, bv_f, nullptr, Vb, MS, DD, DD);
  spec_attn_kernel<<<dim3(NSEQ, 4), blk, lds_spec, stream>>>(Qb, Kb, Vb, Ob);
  gemm_bias_kernel<true><<<gS, blk, 0, stream>>>(Ob, Wh_f, bh_f, x, Qb, MS, DD, DD);
  layernorm128_kernel<<<(MS + 7) / 8, blk, 0, stream>>>(Qb, ga_f, be_f, Kb, MS);
  ffn_kernel<<<MS / 16, blk, lds_ffn, stream>>>(Kb, W1_f, b1_f, W2_f, b2_f, gf_f, bf_f, Vb, MS);

  // ---- temporal transformer on fct token ----
  gather_fct_kernel<<<(NSEQ * 32) / 256, blk, 0, stream>>>(Vb, fct);
  gemm_bias_kernel<false><<<gT, blk, 0, stream>>>(fct, Wq_t, bq_t, nullptr, tQ, NSEQ, DD, DD);
  gemm_bias_kernel<false><<<gT, blk, 0, stream>>>(fct, Wk_t, bk_t, nullptr, tK, NSEQ, DD, DD);
  gemm_bias_kernel<false><<<gT, blk, 0, stream>>>(fct, Wv_t, bv_t, nullptr, tV, NSEQ, DD, DD);
  temp_attn_kernel<<<dim3(32, 8, 4), blk, lds_temp, stream>>>(tQ, tK, tV, pos_k, tO);
  gemm_bias_kernel<true><<<gT, blk, 0, stream>>>(tO, Wh_t, bh_t, fct, tPre, NSEQ, DD, DD);
  layernorm128_kernel<<<NSEQ / 8, blk, 0, stream>>>(tPre, ga_t, be_t, tX1, NSEQ);
  ffn_kernel<<<NSEQ / 16, blk, lds_ffn, stream>>>(tX1, W1_t, b1_t, W2_t, b2_t, gf_t, bf_t, tX2, NSEQ);

  // ---- assemble output ----
  compose_kernel<<<((size_t)NSEQ * FTOK * 32) / 256, blk, 0, stream>>>(Vb, tX2, (float*)d_out);
}
